// QuantizedLinear_85804856639527
// MI455X (gfx1250) — matrix operand order repaired and checked by reading, not compiled, not run on hardware
//
#include <hip/hip_runtime.h>

// QuantizedLinear: y = x @ ((q - zp) * scale).T + bias
// M=8192, K=4096, N=4096. f16 WMMA path (uint8 weights exact in f16).
// Double-buffered LDS staging, register prefetch, NT output stores.

#define MDIM 8192
#define KDIM 4096
#define NDIM 4096
#define NITER (KDIM / 32)

typedef __attribute__((ext_vector_type(16))) _Float16 v16h;
typedef __attribute__((ext_vector_type(8)))  _Float16 v8h;
typedef __attribute__((ext_vector_type(2)))  _Float16 v2h;
typedef __attribute__((ext_vector_type(8)))  float    v8f;
typedef __attribute__((ext_vector_type(4)))  float    v4f;
typedef __attribute__((ext_vector_type(4)))  int      v4i;

// Block tile: 256(M) x 128(N), K-step 32.
// 8 waves arranged 4(M) x 2(N); each wave computes a 64x64 tile = 4x4 WMMA subtiles.
__global__ __launch_bounds__(256) void qlin_wmma_f16(
    const float* __restrict__ X,   // [M,K] fp32
    const int*   __restrict__ Q,   // [N,K] int32 (values 0..255)
    const float* __restrict__ SC,  // [N]
    const float* __restrict__ ZP,  // [N]
    const float* __restrict__ BS,  // [N]
    float*       __restrict__ OUT) // [M,N]
{
    __shared__ __align__(32) _Float16 As[2][256 * 32]; // [m][k] row-major, 2 x 16 KB
    __shared__ __align__(32) _Float16 Bs[2][32 * 128]; // [k][n] transposed, 2 x 8 KB

    const int t       = threadIdx.x;
    const int lane    = t & 31;
    const int wave    = t >> 5;
    const int halfsel = lane >> 4;   // 0 or 1 (lane half)
    const int l15     = lane & 15;
    const int wm      = wave >> 1;   // 0..3
    const int wn      = wave & 1;    // 0..1

    const int blockN = blockIdx.x * 128;
    const int blockM = blockIdx.y * 256;

    // B-staging role: each thread dequantizes 2 consecutive n-rows x 8 k-values.
    const int n0 = (t & 63) * 2;     // 0..126 even
    const int kc = (t >> 6) * 8;     // 0,8,16,24
    const float zp0 = ZP[blockN + n0];
    const float zp1 = ZP[blockN + n0 + 1];
    const int* q0 = Q + (size_t)(blockN + n0) * KDIM;
    const int* q1 = q0 + KDIM;

    // A-staging role: thread t converts row (blockM + t), 32 k-values per step.
    const float* xrow = X + (size_t)(blockM + t) * KDIM;

    // Raw prefetch registers (issued early; consumed in stage()).
    v4f ax[8];
    v4i bx[4];

    auto loadRaw = [&](int k) {
        const float* xs = xrow + k;
#pragma unroll
        for (int c = 0; c < 8; ++c)
            ax[c] = *(const v4f*)(xs + c * 4);
        const int* pa = q0 + k + kc;
        const int* pb = q1 + k + kc;
        bx[0] = *(const v4i*)(pa);
        bx[1] = *(const v4i*)(pa + 4);
        bx[2] = *(const v4i*)(pb);
        bx[3] = *(const v4i*)(pb + 4);
    };

    auto stage = [&](int buf) {
        // A: fp32 -> f16, [m][k] row-major.
        _Float16* Ad = &As[buf][t * 32];
#pragma unroll
        for (int c = 0; c < 4; ++c) {
            v8h o;
#pragma unroll
            for (int i = 0; i < 4; ++i) {
                o[i]     = (_Float16)ax[2 * c][i];
                o[i + 4] = (_Float16)ax[2 * c + 1][i];
            }
            *(v8h*)(Ad + c * 8) = o;
        }
        // B: dequant (q - zp) -> f16, stored TRANSPOSED [k][n].
        int ai[8] = {bx[0][0], bx[0][1], bx[0][2], bx[0][3],
                     bx[1][0], bx[1][1], bx[1][2], bx[1][3]};
        int bi[8] = {bx[2][0], bx[2][1], bx[2][2], bx[2][3],
                     bx[3][0], bx[3][1], bx[3][2], bx[3][3]};
#pragma unroll
        for (int j = 0; j < 8; ++j) {
            v2h p;
            p[0] = (_Float16)((float)ai[j] - zp0);
            p[1] = (_Float16)((float)bi[j] - zp1);
            *(v2h*)(&Bs[buf][(kc + j) * 128 + n0]) = p;  // ds_store_b32
        }
    };

    v8f acc[4][4];
#pragma unroll
    for (int a = 0; a < 4; ++a)
#pragma unroll
        for (int b = 0; b < 4; ++b)
#pragma unroll
            for (int i = 0; i < 8; ++i)
                acc[a][b][i] = 0.0f;

    loadRaw(0);
    stage(0);
    __syncthreads();

    for (int kt = 0; kt < NITER; ++kt) {
        const int cur = kt & 1;

        // Prefetch next tile's raw data (overlaps with the WMMAs below).
        if (kt + 1 < NITER)
            loadRaw((kt + 1) * 32);

        // ---- compute: 4x4 WMMA subtiles per wave ----
        v16h bfrag[4];
#pragma unroll
        for (int sn = 0; sn < 4; ++sn) {
            // B layout: lane holds column n = l15, K = halfsel*8 + {0..7} and +16.
            const _Float16* bp = &Bs[cur][wn * 64 + sn * 16 + l15];
#pragma unroll
            for (int s = 0; s < 16; ++s)
                bfrag[sn][s] = bp[((s < 8) ? (halfsel * 8 + s) : (16 + halfsel * 8 + s - 8)) * 128];
        }
#pragma unroll
        for (int sm = 0; sm < 4; ++sm) {
            // A layout: lane holds row (l15), K = halfsel*8 + {0..7} and +16.
            const _Float16* ap = &As[cur][(wm * 64 + sm * 16 + l15) * 32 + halfsel * 8];
            v8h lo = *(const v8h*)(ap);
            v8h hi = *(const v8h*)(ap + 16);
            v16h af = __builtin_shufflevector(lo, hi,
                0, 1, 2, 3, 4, 5, 6, 7, 8, 9, 10, 11, 12, 13, 14, 15);
#pragma unroll
            for (int sn = 0; sn < 4; ++sn)
                acc[sm][sn] = __builtin_amdgcn_wmma_f32_16x16x32_f16(
                    false, af, false, bfrag[sn],
                    (short)0, acc[sm][sn], false, false);
        }

        // Stage next tile into the other buffer (no conflict with current reads).
        if (kt + 1 < NITER)
            stage(cur ^ 1);
        __syncthreads();
    }

    // ---- epilogue: y = acc*scale + bias; C/D: M = v + 8*halfsel, N = l15 ----
    // Non-temporal stores: keep x/q resident in L2, don't cache the y stream.
#pragma unroll
    for (int sn = 0; sn < 4; ++sn) {
        const int col = blockN + wn * 64 + sn * 16 + l15;
        const float sc = SC[col];
        const float bb = BS[col];
#pragma unroll
        for (int sm = 0; sm < 4; ++sm) {
            float* op = OUT + (size_t)(blockM + wm * 64 + sm * 16 + halfsel * 8) * NDIM + col;
#pragma unroll
            for (int v = 0; v < 8; ++v)
                __builtin_nontemporal_store(acc[sm][sn][v] * sc + bb, op + (size_t)v * NDIM);
        }
    }
}

extern "C" void kernel_launch(void* const* d_in, const int* in_sizes, int n_in,
                              void* d_out, int out_size, void* d_ws, size_t ws_size,
                              hipStream_t stream) {
    (void)in_sizes; (void)n_in; (void)out_size; (void)d_ws; (void)ws_size;
    const float* X  = (const float*)d_in[0];
    const int*   Q  = (const int*)d_in[1];
    const float* SC = (const float*)d_in[2];
    const float* ZP = (const float*)d_in[3];
    const float* BS = (const float*)d_in[4];
    float* OUT = (float*)d_out;

    dim3 grid(NDIM / 128, MDIM / 256);  // 32 x 32 blocks
    dim3 block(256);
    hipLaunchKernelGGL(qlin_wmma_f16, grid, block, 0, stream, X, Q, SC, ZP, BS, OUT);
}
